// MLPPredictor_83021717831905
// MI455X (gfx1250) — matrix operand order repaired and checked by reading, not compiled, not run on hardware
//
#include <hip/hip_runtime.h>

typedef __attribute__((ext_vector_type(16))) __bf16       v16bf;
typedef __attribute__((ext_vector_type(8)))  float        v8f;
typedef __attribute__((ext_vector_type(4)))  unsigned int u32x4;
typedef __attribute__((ext_vector_type(8)))  unsigned int u32x8;

#define KPAD 544   // 32 (7 edge feats + pad) + 256 (h[src]) + 256 (h[dst]) = 17 * 32
#define XSTR 552   // LDS row stride (ushorts) for X tile: 1104B = 276 dw ≡ 20 mod 64 banks
#define HSTR 264   // LDS row stride for hid tile: 528B = 132 dw ≡ 4 mod 64 banks
#define WAVES_PER_BLOCK 2

__device__ __forceinline__ unsigned short f2bf(float f) {
  unsigned int u = __builtin_bit_cast(unsigned int, f);
  u += 0x7FFFu + ((u >> 16) & 1u);            // round-to-nearest-even
  return (unsigned short)(u >> 16);
}

// Repack weights to bf16 in WMMA-B-friendly (N-major, K-contiguous, K-padded) layout.
__global__ void prep_weights(const float* __restrict__ W1,   // [519][256]
                             const float* __restrict__ W2,   // [256][4]
                             unsigned short* __restrict__ w1t,   // [256][KPAD]
                             unsigned short* __restrict__ w2t) { // [16][256]
  int i = blockIdx.x * blockDim.x + threadIdx.x;
  if (i < 256 * KPAD) {
    int n = i / KPAD, k = i % KPAD;
    float v = 0.f;
    if (k < 32) { if (k < 7) v = W1[k * 256 + n]; }        // edge-feat rows 0..6
    else        { v = W1[(k - 25) * 256 + n]; }            // src rows: k-32+7, dst rows: k-288+263 (both k-25)
    w1t[n * KPAD + k] = f2bf(v);
  } else {
    int j = i - 256 * KPAD;
    if (j < 16 * 256) {
      int n = j / 256, k = j % 256;
      float v = (n < 4) ? W2[k * 4 + n] : 0.f;
      w2t[n * 256 + k] = f2bf(v);
    }
  }
}

__global__ void __launch_bounds__(WAVES_PER_BLOCK * 32)
edge_mlp(const float* __restrict__ h,
         const float* __restrict__ ef,
         const int*   __restrict__ src,
         const int*   __restrict__ dst,
         const float* __restrict__ b1,
         const float* __restrict__ b2,
         const unsigned short* __restrict__ w1t,
         const unsigned short* __restrict__ w2t,
         float* __restrict__ out,
         int E) {
  __shared__ __align__(16) unsigned short Xs[WAVES_PER_BLOCK * 16 * XSTR];
  __shared__ __align__(16) unsigned short Hs[WAVES_PER_BLOCK * 16 * HSTR];

  const int lane = threadIdx.x & 31;
  const int wave = threadIdx.x >> 5;
  const int half = lane >> 4;    // which 16-lane half of the wave
  const int mrow = lane & 15;
  const int tile = blockIdx.x * WAVES_PER_BLOCK + wave;
  const int e0   = tile * 16;

  unsigned short* Xw = Xs + wave * 16 * XSTR;
  unsigned short* Hw = Hs + wave * 16 * HSTR;

  // ---------- stage X tile (16 edges x KPAD) as bf16 into LDS ----------
  // cols [0,32): 7 edge features + zero pad
  for (int i = lane; i < 16 * 32; i += 32) {
    int row = i >> 5, col = i & 31;
    int e = e0 + row;
    float v = (col < 7 && e < E) ? ef[(long)e * 7 + col] : 0.f;
    Xw[row * XSTR + col] = f2bf(v);
  }
  // cols [32,288): h[src[e]], cols [288,544): h[dst[e]]  (float4 gathers, L2-resident h)
  for (int pass = 0; pass < 2; ++pass) {
    const int* idxp = pass ? dst : src;
    const int cbase = 32 + pass * 256;
    for (int i = lane; i < 16 * 64; i += 32) {
      int row = i >> 6, q = i & 63;
      int e = e0 + row;
      float4 hv = make_float4(0.f, 0.f, 0.f, 0.f);
      if (e < E) {
        long s = idxp[e];
        hv = *(const float4*)(h + s * 256 + q * 4);
      }
      unsigned int p0 = (unsigned int)f2bf(hv.x) | ((unsigned int)f2bf(hv.y) << 16);
      unsigned int p1 = (unsigned int)f2bf(hv.z) | ((unsigned int)f2bf(hv.w) << 16);
      *(uint2*)(void*)&Xw[row * XSTR + cbase + q * 4] = make_uint2(p0, p1);
    }
  }
  __syncthreads();

  // ---------- layer 1: hid = relu(X @ W1 + b1), 16 N-tiles x 17 K-steps ----------
  const unsigned short* Arow1 = Xw + mrow * XSTR + half * 16;
  for (int nt = 0; nt < 16; ++nt) {
    const int n0 = nt * 16;
    float bias = b1[n0 + mrow];                 // C layout: all 8 VGPRs share N = mrow
    v8f acc = {bias, bias, bias, bias, bias, bias, bias, bias};
    const unsigned short* Brow = w1t + (long)(n0 + mrow) * KPAD + half * 16;
    for (int kk = 0; kk < 17; ++kk) {
      const int k0 = kk * 32;
      // A fragment (16x32 bf16): lane holds row mrow, K = k0 + half*16 .. +15
      u32x4 alo = *(const u32x4*)(const void*)(Arow1 + k0);
      u32x4 ahi = *(const u32x4*)(const void*)(Arow1 + k0 + 8);
      v16bf a = __builtin_bit_cast(v16bf,
                  __builtin_shufflevector(alo, ahi, 0, 1, 2, 3, 4, 5, 6, 7));
      // B fragment (32x16 bf16): lane holds col mrow, K = k0 + half*16 .. +15 (32 contiguous bytes)
      v16bf b = __builtin_bit_cast(v16bf, *(const u32x8*)(const void*)(Brow + k0));
      acc = __builtin_amdgcn_wmma_f32_16x16x32_bf16(false, a, false, b,
                                                    (short)0, acc, false, false);
    }
    // ReLU + cvt, scatter C layout -> row-major hid tile in LDS
#pragma unroll
    for (int r = 0; r < 8; ++r) {
      float v = acc[r] > 0.f ? acc[r] : 0.f;
      Hw[(r + half * 8) * HSTR + n0 + mrow] = f2bf(v);
    }
  }
  __syncthreads();

  // ---------- layer 2: rep = hid @ W2 + b2 (N padded 4 -> 16), 8 K-steps ----------
  {
    float bias = (mrow < 4) ? b2[mrow] : 0.f;
    v8f acc = {bias, bias, bias, bias, bias, bias, bias, bias};
    const unsigned short* Arow = Hw + mrow * HSTR + half * 16;
    const unsigned short* Brow = w2t + mrow * 256 + half * 16;
#pragma unroll
    for (int kk = 0; kk < 8; ++kk) {
      const int k0 = kk * 32;
      u32x4 alo = *(const u32x4*)(const void*)(Arow + k0);
      u32x4 ahi = *(const u32x4*)(const void*)(Arow + k0 + 8);
      v16bf a = __builtin_bit_cast(v16bf,
                  __builtin_shufflevector(alo, ahi, 0, 1, 2, 3, 4, 5, 6, 7));
      v16bf b = __builtin_bit_cast(v16bf, *(const u32x8*)(const void*)(Brow + k0));
      acc = __builtin_amdgcn_wmma_f32_16x16x32_bf16(false, a, false, b,
                                                    (short)0, acc, false, false);
    }
    if (mrow < 4) {
#pragma unroll
      for (int r = 0; r < 8; ++r) {
        int e = e0 + r + half * 8;              // C layout: M = r + half*8
        if (e < E) out[(long)e * 4 + mrow] = acc[r];
      }
    }
  }
}

extern "C" void kernel_launch(void* const* d_in, const int* in_sizes, int n_in,
                              void* d_out, int out_size, void* d_ws, size_t ws_size,
                              hipStream_t stream) {
  const float* h   = (const float*)d_in[0];
  const float* ef  = (const float*)d_in[1];
  const int*   src = (const int*)d_in[2];
  const int*   dst = (const int*)d_in[3];
  const float* W1  = (const float*)d_in[4];
  const float* b1  = (const float*)d_in[5];
  const float* W2  = (const float*)d_in[6];
  const float* b2  = (const float*)d_in[7];
  float* out = (float*)d_out;
  const int E = in_sizes[2];

  unsigned short* w1t = (unsigned short*)d_ws;          // 256*544 bf16 = 278528 B
  unsigned short* w2t = w1t + 256 * KPAD;               // 16*256 bf16 = 8192 B (32B-aligned offset)

  const int prep_elems = 256 * KPAD + 16 * 256;
  prep_weights<<<(prep_elems + 255) / 256, 256, 0, stream>>>(W1, W2, w1t, w2t);

  const int tiles  = (E + 15) / 16;
  const int blocks = (tiles + WAVES_PER_BLOCK - 1) / WAVES_PER_BLOCK;
  edge_mlp<<<blocks, WAVES_PER_BLOCK * 32, 0, stream>>>(h, ef, src, dst, b1, b2,
                                                        w1t, w2t, out, E);
}
